// TSARLayer_41807211659339
// MI455X (gfx1250) — matrix operand order repaired and checked by reading, not compiled, not run on hardware
//
#include <hip/hip_runtime.h>
#include <hip/hip_bf16.h>

typedef __attribute__((ext_vector_type(16))) _Float16 v16h;
typedef __attribute__((ext_vector_type(8)))  float    v8f;

#define EMB     128
#define EA      32
#define ETDIM   32
#define DIN     192          // EMB + EA + ETDIM
#define NNODES  50000
#define NEDGES  500000
#define WPB     8            // waves per 256-thread block (wave32)
#define HALF_TOTAL 3200000u  // NNODES*EMB/2 for threefry counter split

// ---------------- threefry2x32 (exact JAX PRNG for dropout) ----------------
__device__ __forceinline__ unsigned rotl32(unsigned v, int d) {
    return (v << d) | (v >> (32 - d));
}
__device__ __forceinline__ void tf_round(unsigned& x0, unsigned& x1, int d) {
    x0 += x1; x1 = rotl32(x1, d); x1 ^= x0;
}
__device__ __forceinline__ void threefry2x32(unsigned k0, unsigned k1,
                                             unsigned& x0, unsigned& x1) {
    unsigned k2 = k0 ^ k1 ^ 0x1BD11BDAu;
    x0 += k0; x1 += k1;
    tf_round(x0, x1, 13); tf_round(x0, x1, 15); tf_round(x0, x1, 26); tf_round(x0, x1, 6);
    x0 += k1; x1 += k2 + 1u;
    tf_round(x0, x1, 17); tf_round(x0, x1, 29); tf_round(x0, x1, 16); tf_round(x0, x1, 24);
    x0 += k2; x1 += k0 + 2u;
    tf_round(x0, x1, 13); tf_round(x0, x1, 15); tf_round(x0, x1, 26); tf_round(x0, x1, 6);
    x0 += k0; x1 += k1 + 3u;
    tf_round(x0, x1, 17); tf_round(x0, x1, 29); tf_round(x0, x1, 16); tf_round(x0, x1, 24);
    x0 += k1; x1 += k2 + 4u;
    tf_round(x0, x1, 13); tf_round(x0, x1, 15); tf_round(x0, x1, 26); tf_round(x0, x1, 6);
    x0 += k2; x1 += k0 + 5u;
}

// -------- prep: f32 weights -> f16, pre-swizzled into per-lane B-fragment order
// B (32x16, 16-bit): element i of lane holds K=(i/8)*16 + (lane/16)*8 + i%8; col = lane%16.
// dst index = ((kt*8 + nt)*32 + lane)*16 + i  -> one contiguous v16h per fragment.
__global__ void prep_swizzle(const float* __restrict__ W, _Float16* __restrict__ dst, int KT) {
    int tid = blockIdx.x * blockDim.x + threadIdx.x;
    int total = KT * 8 * 32 * 16;
    if (tid >= total) return;
    int i    = tid & 15;
    int lane = (tid >> 4) & 31;
    int nt   = (tid >> 9) & 7;
    int kt   = tid >> 12;
    int k = kt * 32 + (i >> 3) * 16 + (lane >> 4) * 8 + (i & 7);
    int n = nt * 16 + (lane & 15);
    dst[tid] = (_Float16)W[k * 128 + n];
}

// -------- edge kernel: gather + concat -> f16 LDS tile -> WMMA GEMM -> relu -> atomic scatter
__global__ __launch_bounds__(256) void edge_msg_kernel(
    const float* __restrict__ hidden, const int* __restrict__ edge_index,
    const float* __restrict__ eattr,  const float* __restrict__ etemb,
    const float* __restrict__ bmsg,   const _Float16* __restrict__ wmsg16,
    float* __restrict__ accum)
{
    __shared__ _Float16 sm[WPB][16][200];   // 16 edges x 192 feats, padded stride
    const int lane = threadIdx.x & 31;
    const int w    = threadIdx.x >> 5;
    const int tile = blockIdx.x * WPB + w;
    const int e0   = tile * 16;
    if (e0 >= NEDGES) return;

    // cooperative gather of [16 x 192] concat(hidden[src], eattr, etemb), f32 -> f16
    for (int idx = lane; idx < 16 * DIN; idx += 32) {
        int e = idx / DIN;
        int f = idx - e * DIN;
        int eg = e0 + e;
        float v;
        if (f < EMB) {
            int s = edge_index[eg];              // row 0: source nodes
            v = hidden[s * EMB + f];
        } else if (f < EMB + EA) {
            v = eattr[eg * EA + (f - EMB)];
        } else {
            v = etemb[eg * ETDIM + (f - EMB - EA)];
        }
        sm[w][e][f] = (_Float16)v;
    }
    // same-wave LDS ops are in order: no barrier needed (private LDS slice per wave)

    v8f acc[8] = {};
    const int m  = lane & 15;
    const int kh = (lane >> 4) * 8;   // ISA 16-bit A 16x32 layout

    #pragma unroll
    for (int kt = 0; kt < 6; ++kt) {
        v16h a;
        #pragma unroll
        for (int g = 0; g < 2; ++g)
            #pragma unroll
            for (int j = 0; j < 8; ++j)
                a[g * 8 + j] = sm[w][m][kt * 32 + g * 16 + kh + j];
        #pragma unroll
        for (int nt = 0; nt < 8; ++nt) {
            v16h b = *(const v16h*)(wmsg16 + ((kt * 8 + nt) * 32 + lane) * 16);
            acc[nt] = __builtin_amdgcn_wmma_f32_16x16x32_f16(
                false, a, false, b, (short)0, acc[nt], false, false);
        }
    }

    int dsts[8];
    #pragma unroll
    for (int r = 0; r < 8; ++r)
        dsts[r] = edge_index[NEDGES + e0 + (lane >> 4) * 8 + r];  // row 1: dest nodes

    #pragma unroll
    for (int nt = 0; nt < 8; ++nt) {
        int col = nt * 16 + (lane & 15);
        float bias = bmsg[col];
        #pragma unroll
        for (int r = 0; r < 8; ++r) {
            float v = acc[nt][r] + bias;
            v = v > 0.0f ? v : 0.0f;                 // relu on message
            atomicAdd(&accum[(long long)dsts[r] * EMB + col], v);
        }
    }
}

// -------- node kernel: accum @ W_lin + b -> LayerNorm -> relu -> threefry dropout
__global__ __launch_bounds__(256) void node_kernel(
    const float* __restrict__ accum, const _Float16* __restrict__ wlin16,
    const float* __restrict__ blin,  const float* __restrict__ gamma,
    const float* __restrict__ beta,  float* __restrict__ out)
{
    __shared__ _Float16 sm[WPB][16][136];
    const int lane = threadIdx.x & 31;
    const int w    = threadIdx.x >> 5;
    const int tile = blockIdx.x * WPB + w;
    const int n0   = tile * 16;
    if (n0 >= NNODES) return;

    for (int idx = lane; idx < 16 * EMB; idx += 32) {
        int r = idx >> 7;
        int f = idx & 127;
        sm[w][r][f] = (_Float16)accum[(long long)(n0 + r) * EMB + f];
    }

    v8f acc[8] = {};
    const int m  = lane & 15;
    const int kh = (lane >> 4) * 8;

    #pragma unroll
    for (int kt = 0; kt < 4; ++kt) {
        v16h a;
        #pragma unroll
        for (int g = 0; g < 2; ++g)
            #pragma unroll
            for (int j = 0; j < 8; ++j)
                a[g * 8 + j] = sm[w][m][kt * 32 + g * 16 + kh + j];
        #pragma unroll
        for (int nt = 0; nt < 8; ++nt) {
            v16h b = *(const v16h*)(wlin16 + ((kt * 8 + nt) * 32 + lane) * 16);
            acc[nt] = __builtin_amdgcn_wmma_f32_16x16x32_f16(
                false, a, false, b, (short)0, acc[nt], false, false);
        }
    }

    // bias (col fixed per nt/lane across all 8 row-registers)
    #pragma unroll
    for (int nt = 0; nt < 8; ++nt) {
        float bias = blin[nt * 16 + (lane & 15)];
        #pragma unroll
        for (int r = 0; r < 8; ++r) acc[nt][r] += bias;
    }

    // LayerNorm stats: row m lives entirely in one 16-lane half (C layout),
    // so reduce with xor-shuffles of mask 1,2,4,8 (stay within the half).
    float mean[8], rstd[8];
    #pragma unroll
    for (int r = 0; r < 8; ++r) {
        float s = 0.f, sq = 0.f;
        #pragma unroll
        for (int nt = 0; nt < 8; ++nt) { float v = acc[nt][r]; s += v; sq += v * v; }
        #pragma unroll
        for (int off = 1; off < 16; off <<= 1) {
            s  += __shfl_xor(s,  off, 32);
            sq += __shfl_xor(sq, off, 32);
        }
        float mu  = s  * (1.0f / 128.0f);
        float var = sq * (1.0f / 128.0f) - mu * mu;   // biased var, like jnp.var
        mean[r] = mu;
        rstd[r] = rsqrtf(var + 1e-5f);
    }

    const int mlo = (lane >> 4) * 8;
    #pragma unroll
    for (int nt = 0; nt < 8; ++nt) {
        int col = nt * 16 + (lane & 15);
        float g  = gamma[col];
        float bt = beta[col];
        #pragma unroll
        for (int r = 0; r < 8; ++r) {
            float v = (acc[nt][r] - mean[r]) * rstd[r] * g + bt;
            v = v > 0.f ? v : 0.f;
            // dropout p=0.1, exact JAX threefry2x32 with key (0,42)
            unsigned idx = (unsigned)(n0 + mlo + r) * 128u + (unsigned)col;
            unsigned j  = idx < HALF_TOTAL ? idx : idx - HALF_TOTAL;
            unsigned x0 = j, x1 = j + HALF_TOTAL;
            threefry2x32(0u, 42u, x0, x1);
            unsigned bits = idx < HALF_TOTAL ? x0 : x1;
            float u = __uint_as_float((bits >> 9) | 0x3f800000u) - 1.0f;
            v = (u < 0.9f) ? v * (1.0f / 0.9f) : 0.0f;
            out[idx] = v;
        }
    }
}

extern "C" void kernel_launch(void* const* d_in, const int* in_sizes, int n_in,
                              void* d_out, int out_size, void* d_ws, size_t ws_size,
                              hipStream_t stream) {
    (void)in_sizes; (void)n_in; (void)out_size; (void)ws_size;
    const float* hidden = (const float*)d_in[0];
    const int*   eidx   = (const int*)d_in[1];
    const float* eattr  = (const float*)d_in[2];
    const float* etemb  = (const float*)d_in[3];
    const float* bc     = (const float*)d_in[4];
    const float* Wmsg   = (const float*)d_in[5];
    const float* bmsg   = (const float*)d_in[6];
    const float* Wlin   = (const float*)d_in[7];
    const float* blin   = (const float*)d_in[8];
    const float* gamma  = (const float*)d_in[9];
    const float* beta   = (const float*)d_in[10];
    float* out = (float*)d_out;

    char* ws = (char*)d_ws;
    _Float16* wmsg16 = (_Float16*)ws;                     // 192*128*2 = 49152 B
    _Float16* wlin16 = (_Float16*)(ws + 49152);           // 128*128*2 = 32768 B
    float*    accum  = (float*)(ws + 49152 + 32768);      // N*128*4  = 25.6 MB

    // weight conversion + swizzle (tiny)
    prep_swizzle<<<(6 * 8 * 32 * 16 + 255) / 256, 256, 0, stream>>>(Wmsg, wmsg16, 6);
    prep_swizzle<<<(4 * 8 * 32 * 16 + 255) / 256, 256, 0, stream>>>(Wlin, wlin16, 4);

    // accumulator seeded with boundary_condition == the self-loop messages
    hipMemcpyAsync(accum, bc, (size_t)NNODES * EMB * sizeof(float),
                   hipMemcpyDeviceToDevice, stream);

    // edge GEMM + scatter: 31250 waves of 16 edges, 8 waves/block
    int edge_waves  = NEDGES / 16;
    int edge_blocks = (edge_waves + WPB - 1) / WPB;
    edge_msg_kernel<<<edge_blocks, 256, 0, stream>>>(hidden, eidx, eattr, etemb,
                                                     bmsg, wmsg16, accum);

    // node GEMM + LN + relu + dropout: 3125 waves of 16 nodes
    int node_waves  = NNODES / 16;
    int node_blocks = (node_waves + WPB - 1) / WPB;
    node_kernel<<<node_blocks, 256, 0, stream>>>(accum, wlin16, blin, gamma, beta, out);
}
